// encoder_57870389347097
// MI455X (gfx1250) — compile-verified, matrix operand order checked by reading, not run on hardware
//
#include <hip/hip_runtime.h>
#include <hip/hip_bf16.h>
#include <math.h>

// ---------------------------------------------------------------------------
// Types for CDNA5 WMMA (wave32): v_wmma_f32_16x16x32_bf16
// ---------------------------------------------------------------------------
typedef __bf16 bf16_t;
typedef __attribute__((ext_vector_type(16))) __bf16 v16bf;
typedef __attribute__((ext_vector_type(8)))  float  v8f;

#define CEILDIV(a, b) (((a) + (b) - 1) / (b))

__device__ __forceinline__ float siluf(float x)  { return x / (1.f + __expf(-x)); }
__device__ __forceinline__ float geluf(float x)  { return 0.5f * x * (1.f + erff(x * 0.70710678118654752f)); }
__device__ __forceinline__ float softplusf(float x) { return (x > 20.f) ? x : log1pf(__expf(x)); }

// Load 16 contiguous fp32 (16B-aligned) and convert to a bf16 fragment.
// Lowers to 4x global_load_b128 + 8x v_cvt_pk_bf16_f32 -- no per-element guards.
__device__ __forceinline__ v16bf load_frag_f32(const float* __restrict__ p)
{
  const float4* q = (const float4*)p;
  float4 f0 = q[0], f1 = q[1], f2 = q[2], f3 = q[3];
  v16bf r;
  r[0]  = (bf16_t)f0.x; r[1]  = (bf16_t)f0.y; r[2]  = (bf16_t)f0.z; r[3]  = (bf16_t)f0.w;
  r[4]  = (bf16_t)f1.x; r[5]  = (bf16_t)f1.y; r[6]  = (bf16_t)f1.z; r[7]  = (bf16_t)f1.w;
  r[8]  = (bf16_t)f2.x; r[9]  = (bf16_t)f2.y; r[10] = (bf16_t)f2.z; r[11] = (bf16_t)f2.w;
  r[12] = (bf16_t)f3.x; r[13] = (bf16_t)f3.y; r[14] = (bf16_t)f3.z; r[15] = (bf16_t)f3.w;
  return r;
}

// ---------------------------------------------------------------------------
// WMMA GEMM:  C[M,N] = act( A[M,K] * B[N,K]^T + bias[N] )
// Requirements guaranteed by the caller:
//   - M multiple of 64 (grid exact), K multiple of 32
//   - C buffer has ldc >= 16*gridDim.y (pad columns writable scrap)
//   - B row index clamped (branchless) for ragged N; A/B 16B-aligned rows
// Block = 4 waves stacked along M sharing one 16-column B tile staged in LDS.
// act: 0 = none, 1 = softplus
// ---------------------------------------------------------------------------
#define MAX_K 512
__global__ void k_gemm_wmma(const float* __restrict__ A, int lda,
                            const float* __restrict__ B, int ldb,
                            const float* __restrict__ bias,
                            float* __restrict__ C, int ldc,
                            int M, int N, int K, int act)
{
  __shared__ bf16_t Bs[16 * MAX_K];   // 16 KB max

  const int lane  = threadIdx.x;      // 0..31
  const int wave  = threadIdx.y;      // 0..3
  const int l16   = lane & 15;
  const int half  = lane >> 4;        // 0: K lo 16, 1: K hi 16
  const int mBase = (blockIdx.x * 4 + wave) * 16;
  const int nBase = blockIdx.y * 16;

  // ---- cooperative B tile -> LDS (bf16), rows clamped for ragged N ----
  {
    for (int row = wave * 4; row < wave * 4 + 4; ++row) {
      const int gr = min(nBase + row, N - 1);
      const float* src = B + (size_t)gr * ldb;
      for (int k = lane * 4; k < K; k += 128) {
        float4 f = *(const float4*)(src + k);
        bf16_t* d = &Bs[(size_t)row * K + k];
        d[0] = (bf16_t)f.x; d[1] = (bf16_t)f.y; d[2] = (bf16_t)f.z; d[3] = (bf16_t)f.w;
      }
    }
  }
  __syncthreads();

  const float*  aPtr = A + (size_t)(mBase + l16) * lda + half * 16;
  const bf16_t* bPtr = &Bs[(size_t)l16 * K + half * 16];

  v8f acc0 = {}, acc1 = {};
  int k0 = 0;
  for (; k0 + 64 <= K; k0 += 64) {
    __builtin_prefetch(aPtr + k0 + 64, 0, 0);       // gfx1250 global_prefetch_b8
    v16bf a0 = load_frag_f32(aPtr + k0);
    v16bf b0 = *(const v16bf*)(bPtr + k0);          // 2x ds_load_b128
    acc0 = __builtin_amdgcn_wmma_f32_16x16x32_bf16(false, a0, false, b0,
                                                   (short)0, acc0, false, false);
    v16bf a1 = load_frag_f32(aPtr + k0 + 32);
    v16bf b1 = *(const v16bf*)(bPtr + k0 + 32);
    acc1 = __builtin_amdgcn_wmma_f32_16x16x32_bf16(false, a1, false, b1,
                                                   (short)0, acc1, false, false);
  }
  if (k0 < K) {                                      // K % 64 == 32 tail
    v16bf a0 = load_frag_f32(aPtr + k0);
    v16bf b0 = *(const v16bf*)(bPtr + k0);
    acc0 = __builtin_amdgcn_wmma_f32_16x16x32_bf16(false, a0, false, b0,
                                                   (short)0, acc0, false, false);
  }

  // C/D layout: VGPR v -> row = v + 8*half, col = lane%16.  Stores unconditional.
  const int c = nBase + l16;
  const float bv = bias ? bias[min(c, N - 1)] : 0.f;
#pragma unroll
  for (int v = 0; v < 8; ++v) {
    const int r = mBase + v + half * 8;
    float o = acc0[v] + acc1[v] + bv;
    if (act == 1) o = softplusf(o);
    C[(size_t)r * ldc + c] = o;
  }
}

// ---------------------------------------------------------------------------
// Zero-pad dt_w [di,r] -> [di,32] so the dt GEMM runs guard-free with K=32.
// ---------------------------------------------------------------------------
__global__ void k_pad_dtw(const float* __restrict__ w, float* __restrict__ out,
                          int di, int r)
{
  int idx = blockIdx.x * blockDim.x + threadIdx.x;
  if (idx >= di * 32) return;
  int k = idx & 31, d = idx >> 5;
  out[idx] = (k < r) ? w[(size_t)d * r + k] : 0.f;
}

// ---------------------------------------------------------------------------
// Layout transposes
// ---------------------------------------------------------------------------
__global__ void k_nchw_to_nhwc(const float* __restrict__ in, float* __restrict__ out,
                               int C, int HW)
{
  int idx = blockIdx.x * blockDim.x + threadIdx.x;
  if (idx >= C * HW) return;
  int c = idx / HW, n = idx % HW;
  out[(size_t)n * C + c] = in[idx];
}

__global__ void k_nhwc_to_nchw(const float* __restrict__ in, float* __restrict__ out,
                               int C, int HW)
{
  int idx = blockIdx.x * blockDim.x + threadIdx.x;
  if (idx >= C * HW) return;
  int c = idx / HW, n = idx % HW;
  out[idx] = in[(size_t)n * C + c];
}

// ---------------------------------------------------------------------------
// 3x3 stride-2 conv (NHWC in, NHWC out) + BatchNorm + exact GELU
// ---------------------------------------------------------------------------
__global__ void k_conv_bn_gelu(const float* __restrict__ in,   // [Hin,Win,Cin]
                               const float* __restrict__ w,    // [D,Cin,3,3]
                               const float* __restrict__ b,
                               const float* __restrict__ g,  const float* __restrict__ be,
                               const float* __restrict__ mu, const float* __restrict__ var,
                               float* __restrict__ out,         // [Hout,Wout,D]
                               int Hin, int Win, int Cin, int Hout, int Wout, int D)
{
  int idx = blockIdx.x * blockDim.x + threadIdx.x;
  if (idx >= Hout * Wout * D) return;
  int oc = idx % D, n = idx / D;
  int ox = n % Wout, oy = n / Wout;
  float acc = b[oc];
  for (int ky = 0; ky < 3; ++ky) {
    int iy = oy * 2 + ky - 1;
    if (iy < 0 || iy >= Hin) continue;
    for (int kx = 0; kx < 3; ++kx) {
      int ix = ox * 2 + kx - 1;
      if (ix < 0 || ix >= Win) continue;
      const float* ip = in + ((size_t)iy * Win + ix) * Cin;
      const float* wp = w + (size_t)oc * Cin * 9 + ky * 3 + kx;
      for (int c = 0; c < Cin; ++c) acc += ip[c] * wp[(size_t)c * 9];
    }
  }
  float xn = (acc - mu[oc]) * rsqrtf(var[oc] + 1e-5f) * g[oc] + be[oc];
  out[(size_t)n * D + oc] = geluf(xn);
}

// ---------------------------------------------------------------------------
// 3x3 neighborhood attention core (per pixel x head), reads fused qkv [HW,3D]
// ---------------------------------------------------------------------------
__global__ void k_natten(const float* __restrict__ qkv, const float* __restrict__ rpb,
                         float* __restrict__ out, int H, int W, int D, int h)
{
  int idx = blockIdx.x * blockDim.x + threadIdx.x;
  int HW = H * W;
  if (idx >= HW * h) return;
  int head = idx % h, n = idx / h;
  int i = n / W, j = n % W;
  int hd = D / h;
  float scale = rsqrtf((float)hd);
  int si = min(max(i - 1, 0), H - 3);
  int sj = min(max(j - 1, 0), W - 3);
  const float* qp = qkv + (size_t)n * 3 * D + head * hd;        // t=0
  float logits[9];
  float mx = -1e30f;
  for (int x = 0; x < 3; ++x)
    for (int y = 0; y < 3; ++y) {
      int ni = si + x, nj = sj + y;
      const float* kp = qkv + ((size_t)ni * W + nj) * 3 * D + D + head * hd;  // t=1
      float s = 0.f;
      for (int e = 0; e < hd; ++e) s += qp[e] * kp[e];
      s *= scale;
      int bi = x + (i - si), bj = y + (j - sj);
      s += rpb[((size_t)head * 5 + bi) * 5 + bj];
      logits[x * 3 + y] = s;
      mx = fmaxf(mx, s);
    }
  float se = 0.f;
  for (int t = 0; t < 9; ++t) { logits[t] = __expf(logits[t] - mx); se += logits[t]; }
  float inv = 1.f / se;
  for (int e = 0; e < hd; ++e) {
    float acc = 0.f;
    for (int x = 0; x < 3; ++x)
      for (int y = 0; y < 3; ++y) {
        int ni = si + x, nj = sj + y;
        acc += logits[x * 3 + y] *
               qkv[((size_t)ni * W + nj) * 3 * D + 2 * D + head * hd + e];   // t=2
      }
    out[(size_t)n * D + head * hd + e] = acc * inv;
  }
}

// ---------------------------------------------------------------------------
// Depthwise causal conv1d (width 4, left pad 3) + SiLU.  xi = xz[:, :di],
// row stride 2*di.
// ---------------------------------------------------------------------------
__global__ void k_dwconv_silu(const float* __restrict__ xz, const float* __restrict__ w,
                              const float* __restrict__ b, float* __restrict__ xc,
                              int L, int di)
{
  int idx = blockIdx.x * blockDim.x + threadIdx.x;
  if (idx >= L * di) return;
  int d = idx % di, l = idx / di;
  float acc = b[d];
#pragma unroll
  for (int t = 0; t < 4; ++t) {
    int ll = l - 3 + t;
    if (ll >= 0) acc += w[(size_t)d * 4 + t] * xz[(size_t)ll * 2 * di + d];
  }
  xc[(size_t)l * di + d] = siluf(acc);
}

// ---------------------------------------------------------------------------
// Selective scan: one thread per channel d, 16 SSM states in registers.
// xdbl rows have padded stride ldx; fused epilogue y = (scan + D*xc) * silu(z)
// ---------------------------------------------------------------------------
__global__ void k_scan(const float* __restrict__ dt, const float* __restrict__ xdbl,
                       const float* __restrict__ xc, const float* __restrict__ A_log,
                       const float* __restrict__ Dp, const float* __restrict__ xz,
                       float* __restrict__ y, int L, int di, int r, int ldx)
{
  int d = blockIdx.x * blockDim.x + threadIdx.x;
  if (d >= di) return;
  float A[16], h[16];
#pragma unroll
  for (int s = 0; s < 16; ++s) { A[s] = -__expf(A_log[(size_t)d * 16 + s]); h[s] = 0.f; }
  const float Dd = Dp[d];
  for (int l = 0; l < L; ++l) {
    const float dtl = dt[(size_t)l * di + d];
    const float xcl = xc[(size_t)l * di + d];
    const float* Bm = xdbl + (size_t)l * ldx + r;
    const float* Cm = Bm + 16;
    float acc = 0.f;
#pragma unroll
    for (int s = 0; s < 16; ++s) {
      h[s] = __expf(dtl * A[s]) * h[s] + dtl * Bm[s] * xcl;
      acc += h[s] * Cm[s];
    }
    const float z = xz[(size_t)l * 2 * di + di + d];
    y[(size_t)l * di + d] = (acc + Dd * xcl) * siluf(z);
  }
}

// ---------------------------------------------------------------------------
// RMSNorm over channel dim (row-per-thread)
// ---------------------------------------------------------------------------
__global__ void k_rms(const float* __restrict__ in, const float* __restrict__ w,
                      float* __restrict__ out, int L, int D)
{
  int l = blockIdx.x * blockDim.x + threadIdx.x;
  if (l >= L) return;
  const float* row = in + (size_t)l * D;
  float ss = 0.f;
  for (int c = 0; c < D; ++c) { float v = row[c]; ss += v * v; }
  float sc = rsqrtf(ss / D + 1e-5f);
  for (int c = 0; c < D; ++c) out[(size_t)l * D + c] = row[c] * sc * w[c];
}

// ---------------------------------------------------------------------------
// Channel concat [xa | xm] -> xo [L, 2D]
// ---------------------------------------------------------------------------
__global__ void k_concat(const float* __restrict__ a, const float* __restrict__ b,
                         float* __restrict__ o, int L, int D)
{
  int idx = blockIdx.x * blockDim.x + threadIdx.x;
  if (idx >= L * 2 * D) return;
  int c = idx % (2 * D), l = idx / (2 * D);
  o[idx] = (c < D) ? a[(size_t)l * D + c] : b[(size_t)l * D + (c - D)];
}

// ---------------------------------------------------------------------------
// Host driver
// ---------------------------------------------------------------------------
extern "C" void kernel_launch(void* const* d_in, const int* in_sizes, int n_in,
                              void* d_out, int out_size, void* d_ws, size_t ws_size,
                              hipStream_t stream)
{
  (void)in_sizes; (void)n_in; (void)out_size; (void)ws_size;
  static const int DIMS[4]  = {32, 64, 128, 256};
  static const int HEADS[4] = {4, 4, 8, 8};
  static const int CINS[4]  = {3, 64, 128, 256};

  int p = 0;
  const float* x = (const float*)d_in[p++];
  // 32 leaves per stage, insertion order:
  //  0 conv_w  1 conv_b  2 bn_g  3 bn_b  4 bn_m  5 bn_v  6 qkv_w  7 qkv_b
  //  8 att_proj_w  9 att_proj_b  10 rpb
  //  11..19 mamba0: in_proj, conv_w, conv_b, x_proj, dt_w, dt_b, A_log, D, out_proj
  //  20..28 mamba1: (same)
  //  29 rms_w  30 c1_w  31 c1_b
  const float* P[4][32];
  for (int s = 0; s < 4; ++s)
    for (int t = 0; t < 32; ++t) P[s][t] = (const float*)d_in[p++];

  float* ws = (float*)d_ws;
  const size_t slotSize = (size_t)16384 * 64;          // 1M floats per ping-pong slot
  float* slotA   = ws;
  float* slotB   = ws + slotSize;
  float* scratch = ws + 2 * slotSize;

  // Stage-0 input: NCHW (3,256,256) -> NHWC
  k_nchw_to_nhwc<<<CEILDIV(3 * 256 * 256, 256), 256, 0, stream>>>(x, slotA, 3, 256 * 256);

  float* cur = slotA;
  float* nxt = slotB;
  int Hin = 256;
  const dim3 gb(32, 4);  // 4 waves per block along M

  for (int s = 0; s < 4; ++s) {
    const int D = DIMS[s], h = HEADS[s], Cin = CINS[s];
    const int Hout = Hin / 2, W = Hout;
    const int L = Hout * W;                // multiple of 64 at every stage
    const int di = 2 * D;
    const int r = D / 16;                  // 2,4,8,16
    const int xdl  = r + 32;               // true x_proj N (34..48)
    const int xdlp = 48;                   // padded row stride (3 N-tiles)

    // scratch arena (rebuilt each stage); all offsets 16B-aligned
    float* a0   = scratch;
    float* qkv  = a0   + (size_t)L * D;
    float* nat  = qkv  + (size_t)L * 3 * D;
    float* xa   = nat  + (size_t)L * D;
    float* xz   = xa   + (size_t)L * D;
    float* xc   = xz   + (size_t)L * 4 * D;
    float* xdbl = xc   + (size_t)L * di;
    float* dt   = xdbl + (size_t)L * xdlp;
    float* yb   = dt   + (size_t)L * di;
    float* mo0  = yb   + (size_t)L * di;
    float* mo1  = mo0  + (size_t)L * D;
    float* xm   = mo1  + (size_t)L * D;
    float* xo   = xm   + (size_t)L * D;
    float* dtwp = xo   + (size_t)L * 2 * D;    // [di,32] zero-padded dt weights

    // conv + BN + GELU
    k_conv_bn_gelu<<<CEILDIV(L * D, 256), 256, 0, stream>>>(
        cur, P[s][0], P[s][1], P[s][2], P[s][3], P[s][4], P[s][5],
        a0, Hin, Hin, Cin, Hout, W, D);

    // fused qkv projection (WMMA)
    k_gemm_wmma<<<dim3(L / 64, (3 * D) / 16), gb, 0, stream>>>(
        a0, D, P[s][6], D, P[s][7], qkv, 3 * D, L, 3 * D, D, 0);

    // neighborhood attention core
    k_natten<<<CEILDIV(L * h, 256), 256, 0, stream>>>(qkv, P[s][10], nat, Hout, W, D, h);

    // attention output projection (WMMA)
    k_gemm_wmma<<<dim3(L / 64, D / 16), gb, 0, stream>>>(
        nat, D, P[s][8], D, P[s][9], xa, D, L, D, D, 0);

    // 2x Mamba blocks
    const float* mi = xa;
    float* mouts[2] = {mo0, mo1};
    for (int mb = 0; mb < 2; ++mb) {
      const float* const* MP = &P[s][11 + mb * 9];
      // in_proj: [L,D] x [4D,D]^T  (WMMA)
      k_gemm_wmma<<<dim3(L / 64, (4 * D) / 16), gb, 0, stream>>>(
          mi, D, MP[0], D, nullptr, xz, 4 * D, L, 4 * D, D, 0);
      // depthwise causal conv + SiLU
      k_dwconv_silu<<<CEILDIV(L * di, 256), 256, 0, stream>>>(xz, MP[1], MP[2], xc, L, di);
      // x_proj: [L,di] x [xdl,di]^T  (WMMA; ragged N -> clamped B rows, padded C)
      k_gemm_wmma<<<dim3(L / 64, CEILDIV(xdl, 16)), gb, 0, stream>>>(
          xc, di, MP[3], di, nullptr, xdbl, xdlp, L, xdl, di, 0);
      // dt: softplus(xdbl[:, :32] x dtw_pad^T + dt_b)  (WMMA, zero-padded K=32)
      k_pad_dtw<<<CEILDIV(di * 32, 256), 256, 0, stream>>>(MP[4], dtwp, di, r);
      k_gemm_wmma<<<dim3(L / 64, di / 16), gb, 0, stream>>>(
          xdbl, xdlp, dtwp, 32, MP[5], dt, di, L, di, 32, 1);
      // selective scan (+ D*xc, * silu(z))
      k_scan<<<CEILDIV(di, 64), 64, 0, stream>>>(dt, xdbl, xc, MP[6], MP[7], xz,
                                                 yb, L, di, r, xdlp);
      // out_proj: [L,di] x [D,di]^T  (WMMA)
      k_gemm_wmma<<<dim3(L / 64, D / 16), gb, 0, stream>>>(
          yb, di, MP[8], di, nullptr, mouts[mb], D, L, D, di, 0);
      mi = mouts[mb];
    }

    // RMSNorm
    k_rms<<<CEILDIV(L, 256), 256, 0, stream>>>(mo1, P[s][29], xm, L, D);
    // concat [xa | xm]
    k_concat<<<CEILDIV(L * 2 * D, 256), 256, 0, stream>>>(xa, xm, xo, L, D);
    // c1 channel mixer: [L,2D] x [2D,2D]^T + b  (WMMA) -> next-stage input (NHWC)
    k_gemm_wmma<<<dim3(L / 64, (2 * D) / 16), gb, 0, stream>>>(
        xo, 2 * D, P[s][30], 2 * D, P[s][31], nxt, 2 * D, L, 2 * D, 2 * D, 0);

    // emit stage outputs 1..3 (NHWC -> NCHW, flat-concatenated in d_out)
    if (s >= 1) {
      const size_t off = (s == 1) ? 0u : (s == 2) ? 524288u : 786432u;
      k_nhwc_to_nchw<<<CEILDIV(L * 2 * D, 256), 256, 0, stream>>>(
          nxt, (float*)d_out + off, 2 * D, L);
    }

    // ping-pong
    float* t = cur; cur = nxt; nxt = t;
    Hin = Hout;
  }
}
